// ExtendedAtomEncoder_75866302317033
// MI455X (gfx1250) — compile-verified, matrix operand order checked by reading, not run on hardware
//
#include <hip/hip_runtime.h>
#include <hip/hip_bf16.h>

typedef __attribute__((ext_vector_type(16))) __bf16 v16bf;
typedef __attribute__((ext_vector_type(8)))  float  v8f;

#define BATCH     128
#define MAXN      1024
#define DIM       256
#define ROWS_BLK  256          // M rows per block (8 waves x 2 tiles x 16 rows)
#define LDS_STRIDE 264         // bf16 elems per LDS row (528 B, 16B-aligned, bank-skewed)

// ---------------- prep 1: W[:,0:256] -> bf16 ----------------
__global__ __launch_bounds__(256) void conv_w_kernel(const float* __restrict__ lin_w,
                                                     __bf16* __restrict__ wbf) {
    const int o = blockIdx.x;          // 0..255 output channel
    const int d = threadIdx.x;         // 0..255 enc dim
    wbf[o * DIM + d] = (__bf16)lin_w[o * (2 * DIM) + d];
}

// ---------------- prep 2: per-batch fused bias ----------------
// bias2[b,o] = lin_b[o] + sum_d rxn_emb[rxn_class[b],d] * lin_w[o, 256+d]
__global__ __launch_bounds__(256) void bias_kernel(const int* __restrict__ rxn_class,
                                                   const float* __restrict__ rxn_emb,
                                                   const float* __restrict__ lin_w,
                                                   const float* __restrict__ lin_b,
                                                   float* __restrict__ bias2) {
    __shared__ float cls[DIM];
    const int b = blockIdx.x;
    const int o = threadIdx.x;
    cls[o] = rxn_emb[rxn_class[b] * DIM + o];
    __syncthreads();
    float acc = lin_b[o];
    const float* wrow = lin_w + o * (2 * DIM) + DIM;
    #pragma unroll 4
    for (int d = 0; d < DIM; ++d) acc = fmaf(cls[d], wrow[d], acc);
    bias2[b * DIM + o] = acc;
}

// ---------------- main: gather-sum encoder + bf16 WMMA GEMM ----------------
__global__ __launch_bounds__(256) void atom_encoder_gemm_kernel(
    const int*   __restrict__ node_feat,   // [B*1024*9]
    const int*   __restrict__ num_nodes,   // [B]
    const float* __restrict__ atom_emb,    // [174*256]
    const __bf16* __restrict__ wbf,        // [256*256] bf16 (row o major)
    const float* __restrict__ bias2,       // [B*256]
    float*       __restrict__ out)         // [B*1024*256]
{
    __shared__ __bf16 sA[ROWS_BLK * LDS_STRIDE];   // 135168 B

    const int tid     = threadIdx.x;
    const int rowBase = blockIdx.x * ROWS_BLK;     // 1024/256=4 blocks per batch -> single batch/block
    const int b       = rowBase >> 10;

    // -------- Phase 1: enc rows (sum of 9 embedding gathers), masked, -> LDS bf16 --------
    {
        const int r = tid;                         // one row per thread
        const int m = rowBase + r;
        const int n = m & (MAXN - 1);
        const bool valid = n < num_nodes[b];

        int idx[9];
        if (valid) {
            const int off[9] = {0, 119, 124, 136, 148, 158, 164, 170, 172};
            #pragma unroll
            for (int f = 0; f < 9; ++f)
                idx[f] = (node_feat[m * 9 + f] + off[f]) * DIM;
        }
        __bf16* dst = &sA[r * LDS_STRIDE];
        #pragma unroll 2
        for (int c = 0; c < DIM; c += 4) {
            float sx = 0.f, sy = 0.f, sz = 0.f, sw = 0.f;
            if (valid) {
                #pragma unroll
                for (int f = 0; f < 9; ++f) {
                    const float4 e = *(const float4*)(atom_emb + idx[f] + c);
                    sx += e.x; sy += e.y; sz += e.z; sw += e.w;
                }
            }
            union { __bf16 h[4]; uint2 u; } pk;
            pk.h[0] = (__bf16)sx; pk.h[1] = (__bf16)sy;
            pk.h[2] = (__bf16)sz; pk.h[3] = (__bf16)sw;
            *(uint2*)(dst + c) = pk.u;             // ds_store_b64
        }
    }
    __syncthreads();

    // -------- Phase 2: per-wave 32(M) x 256(N) tile, K=256; each B frag feeds 2 WMMAs --------
    const int lane = tid & 31;
    const int wv   = tid >> 5;        // wave id 0..7 -> 32-row M super-tile
    const int ln   = lane & 15;       // N column within tile / A row
    const int hi   = lane >> 4;       // K-half selector (A/B), M-half selector (C/D)

    union Frag { v16bf v; uint4 u[2]; };

    // A fragments for two 16-row tiles, K = 0..255 (held in registers, 128 VGPRs)
    Frag a0[8], a1[8];
    {
        const __bf16* ar0 = &sA[(wv * 32 + ln) * LDS_STRIDE + hi * 8];
        const __bf16* ar1 = ar0 + 16 * LDS_STRIDE;
        #pragma unroll
        for (int ks = 0; ks < 8; ++ks) {
            a0[ks].u[0] = *(const uint4*)(ar0 + ks * 32);        // ds_load_b128
            a0[ks].u[1] = *(const uint4*)(ar0 + ks * 32 + 16);
            a1[ks].u[0] = *(const uint4*)(ar1 + ks * 32);
            a1[ks].u[1] = *(const uint4*)(ar1 + ks * 32 + 16);
        }
    }

    #pragma unroll 1
    for (int nt = 0; nt < 16; ++nt) {
        v8f acc0 = {};
        v8f acc1 = {};
        const __bf16* wp = wbf + (nt * 16 + ln) * DIM + hi * 8;  // column N=o of W^T = row o of W
        #pragma unroll
        for (int ks = 0; ks < 8; ++ks) {
            Frag bf;
            bf.u[0] = *(const uint4*)(wp + ks * 32);             // global_load_b128 (L2-resident)
            bf.u[1] = *(const uint4*)(wp + ks * 32 + 16);
            acc0 = __builtin_amdgcn_wmma_f32_16x16x32_bf16(
                false, a0[ks].v, false, bf.v, (short)0, acc0, false, false);
            acc1 = __builtin_amdgcn_wmma_f32_16x16x32_bf16(
                false, a1[ks].v, false, bf.v, (short)0, acc1, false, false);
        }
        const float bias = bias2[b * DIM + nt * 16 + ln];
        // C/D layout: VGPR r -> M = hi*8 + r, N = ln
        float* op0 = out + (size_t)(rowBase + wv * 32 + hi * 8) * DIM + nt * 16 + ln;
        float* op1 = op0 + (size_t)16 * DIM;
        #pragma unroll
        for (int r = 0; r < 8; ++r) {
            op0[(size_t)r * DIM] = acc0[r] + bias;
            op1[(size_t)r * DIM] = acc1[r] + bias;
        }
    }
}

extern "C" void kernel_launch(void* const* d_in, const int* in_sizes, int n_in,
                              void* d_out, int out_size, void* d_ws, size_t ws_size,
                              hipStream_t stream) {
    const int*   node_feat = (const int*)  d_in[0];   // [128*1024*9]
    const int*   num_nodes = (const int*)  d_in[1];   // [128]
    const int*   rxn_class = (const int*)  d_in[2];   // [128]
    const float* atom_emb  = (const float*)d_in[3];   // [174*256]
    const float* rxn_emb   = (const float*)d_in[4];   // [10*256]
    const float* lin_w     = (const float*)d_in[5];   // [256*512]
    const float* lin_b     = (const float*)d_in[6];   // [256]
    float*       out       = (float*)d_out;           // [128*1024*256]

    __bf16* wbf   = (__bf16*)d_ws;                          // 256*256*2   = 128 KB
    float*  bias2 = (float*)((char*)d_ws + DIM * DIM * 2);  // 128*256*4   = 128 KB

    conv_w_kernel<<<DIM, 256, 0, stream>>>(lin_w, wbf);
    bias_kernel<<<BATCH, 256, 0, stream>>>(rxn_class, rxn_emb, lin_w, lin_b, bias2);

    const int nblocks = (BATCH * MAXN) / ROWS_BLK;          // 512
    atom_encoder_gemm_kernel<<<nblocks, 256, 0, stream>>>(
        node_feat, num_nodes, atom_emb, wbf, bias2, out);
}
